// LCAOOut_12438225289736
// MI455X (gfx1250) — matrix operand order repaired and checked by reading, not compiled, not run on hardware
//
#include <hip/hip_runtime.h>
#include <stdint.h>

// ---------------------------------------------------------------------------
// MI455X / gfx1250: bf16 WMMA implementation of the LCAO output heads.
//   prop  = sum_n MLP(x)                       (128->128->64->1, SiLU)
//   force = scatter_s( MLP(concat(x[s],x[t])) * edge_vec )  (256->128->64->1)
// Weights pre-packed to bf16 WMMA-B fragment order in d_ws, staged in LDS via
// async-to-LDS; x pre-converted to bf16 (if ws_size allows) so gathered A
// fragments need no packing; per-wave 16-row tiles drive
// v_wmma_f32_16x16x32_bf16 with f32 accumulation.
// ---------------------------------------------------------------------------

typedef __attribute__((ext_vector_type(16))) __bf16 v16bf;
typedef __attribute__((ext_vector_type(8)))  float  v8f;
typedef __attribute__((ext_vector_type(4)))  int    v4i;

#define EMB    128
#define NNODES 50000
#define NEDGES 800000

// d_ws layout:
//   uint16[0      .. 16383]  w1p  : W1  (128x128) fragment-major bf16
//   uint16[16384  .. 24575]  w2p  : W2  (128x64)
//   uint16[24576  .. 57343]  wf1p : Wf1 (256x128)
//   uint16[57344  .. 65535]  wf2p : Wf2 (128x64)
//   uint16[65536  ..      ]  xbf  : x as bf16, row-major (N x 128)  [optional]

// ---- fast f32x2 -> packed bf16x2 -------------------------------------------
__device__ __forceinline__ uint32_t pack_bf16(float lo, float hi) {
  // round-bias + single byte-permute: dst = {hi[31:16], lo[31:16]}
  uint32_t a = __float_as_uint(lo) + 0x8000u;
  uint32_t b = __float_as_uint(hi) + 0x8000u;
  return __builtin_amdgcn_perm(b, a, 0x07060302);
}

__device__ __forceinline__ uint16_t f2bf(float f) {        // RNE (prep only)
  uint32_t u = __float_as_uint(f);
  u += 0x7FFFu + ((u >> 16) & 1u);
  return (uint16_t)(u >> 16);
}

__device__ __forceinline__ uint16_t f2bf_fast(float f) {   // hot loop
  return (uint16_t)((__float_as_uint(f) + 0x8000u) >> 16);
}

__device__ __forceinline__ void lds_fence() {
  asm volatile("s_wait_dscnt 0" ::: "memory");
}

__device__ __forceinline__ float silu(float v) {
  return v / (1.0f + __expf(-v));
}

// ---- fragment container ----------------------------------------------------
union BU {
  uint32_t u[8];
  uint4    q[2];
  v16bf    v;
};

// B fragment from LDS: one lane = 32B contiguous -> 2x ds_load_b128
__device__ __forceinline__ void loadB(BU& B, const uint4* ldsW, int tileBase,
                                      int lane) {
  const uint4* bp = &ldsW[tileBase + lane * 2];
  B.q[0] = bp[0];
  B.q[1] = bp[1];
}

// A-fragment k index for VGPR word v (ISA 7.12.2, 16-bit A 16x32):
// lane half h: k = (v>>2)*16 + (v&3)*2 + 8*h  (pair k, k+1)
__device__ __forceinline__ int a_kk(int v, int half) {
  return ((v >> 2) << 4) + ((v & 3) << 1) + (half << 3);
}

// A fragment straight from a bf16 row (no packing): 4x 8B loads
__device__ __forceinline__ void loadA_bf(BU& A, const uint16_t* rowbf, int kb,
                                         int half) {
  const uint2* p = (const uint2*)((const uint32_t*)rowbf + (kb >> 1) + half * 4);
  uint2 a = p[0], b = p[1];       // dwords for words 0..3 (K = kb+8h .. +7)
  uint2 cq = p[4], d = p[5];      // dwords for words 4..7 (K = kb+16+8h ..)
  A.q[0] = make_uint4(a.x, a.y, b.x, b.y);
  A.q[1] = make_uint4(cq.x, cq.y, d.x, d.y);
}

// A fragment from an f32 row with on-the-fly packing (fallback)
__device__ __forceinline__ void loadA_f32(BU& A, const float* rp, int kb,
                                          int half) {
#pragma unroll
  for (int vv = 0; vv < 8; ++vv) {
    float2 f2 = *(const float2*)(rp + kb + a_kk(vv, half));
    A.u[vv] = pack_bf16(f2.x, f2.y);
  }
}

// ---- cooperative weight staging: global -> LDS (async path) ----------------
__device__ __forceinline__ void stage_weights(const uint4* __restrict__ g,
                                              uint4* l, int n) {
#if __has_builtin(__builtin_amdgcn_global_load_async_to_lds_b128)
  for (int i = threadIdx.x; i < n; i += 256)
    __builtin_amdgcn_global_load_async_to_lds_b128((v4i*)(g + i), (v4i*)(l + i),
                                                   0, 0);
#if __has_builtin(__builtin_amdgcn_s_wait_asynccnt)
  __builtin_amdgcn_s_wait_asynccnt(0);
#else
  asm volatile("s_wait_asynccnt 0" ::: "memory");
#endif
#else
  for (int i = threadIdx.x; i < n; i += 256) l[i] = g[i];
#endif
  __syncthreads();
}

// --------------------------- utility kernels --------------------------------
__global__ void zero_out_kernel(float* out, int n) {
  int i = blockIdx.x * 256 + threadIdx.x;
  if (i < n) out[i] = 0.0f;
}

__global__ void xcvt_kernel(const float* __restrict__ x,
                            uint32_t* __restrict__ xb, int n2) {
  int i = blockIdx.x * 256 + threadIdx.x;
  if (i < n2) {
    float2 f2 = ((const float2*)x)[i];
    xb[i] = pack_bf16(f2.x, f2.y);
  }
}

// ------------------- weight repack: f32 -> bf16 fragment-major --------------
// B-fragment layout (ISA 7.12.2, 16-bit B 32x16): lane = half*16 + c holds
// column n=c; element r holds k = ks*32 + (r>>3)*16 + half*8 + (r&7).
__global__ void prep_weights_kernel(const float* __restrict__ W1,
                                    const float* __restrict__ W2,
                                    const float* __restrict__ Wf1,
                                    const float* __restrict__ Wf2,
                                    uint16_t* __restrict__ ws) {
  int gid = blockIdx.x * 256 + threadIdx.x;
  const float* src;
  int N, dstOff, lid;
  if      (gid < 16384) { src = W1;  N = 128; dstOff = 0;     lid = gid;         }
  else if (gid < 24576) { src = W2;  N = 64;  dstOff = 16384; lid = gid - 16384; }
  else if (gid < 57344) { src = Wf1; N = 128; dstOff = 24576; lid = gid - 24576; }
  else if (gid < 65536) { src = Wf2; N = 64;  dstOff = 57344; lid = gid - 57344; }
  else return;
  int numNt = N >> 4;
  int tile = lid >> 9;            // 512 bf16 elements per 32x16 tile
  int e    = lid & 511;
  int lane = e >> 4;
  int r    = e & 15;
  int v    = r >> 1, p = r & 1;
  int half = lane >> 4, c = lane & 15;
  int ks   = tile / numNt, nt = tile % numNt;
  int k = ks * 32 + (r >> 3) * 16 + half * 8 + (r & 7);
  int n = nt * 16 + c;
  ws[dstOff + lid] = f2bf(src[(size_t)k * N + n]);
}

// ------------------------------ property head ------------------------------
template <bool XBF>
__global__ __launch_bounds__(256) void prop_kernel(
    const float* __restrict__ x, const uint16_t* __restrict__ xbf,
    const float* __restrict__ b1, const float* __restrict__ b2,
    const float* __restrict__ W3, const uint16_t* __restrict__ wblob,
    float* __restrict__ out) {
  __shared__ uint4 ldsW[3072];                          // 48KB: w1p + w2p
  __shared__ __align__(16) uint16_t h1s[8 * 16 * EMB];  // 32KB per-wave staging

  stage_weights((const uint4*)wblob, ldsW, 3072);

  int lane = threadIdx.x & 31;
  int wave = threadIdx.x >> 5;
  int half = lane >> 4;
  int c    = lane & 15;
  int tile = blockIdx.x * 8 + wave;
  if (tile >= NNODES / 16) return;

  size_t node = (size_t)(tile * 16 + c);
  const float*    rowf = x   + node * EMB;
  const uint16_t* rowb = xbf + node * EMB;

  // ---- layer 1: (16x128) @ (128x128) ----
  v8f acc[8] = {};
#pragma unroll
  for (int ks = 0; ks < 4; ++ks) {
    BU A;
    if (XBF) loadA_bf(A, rowb, ks * 32, half);
    else     loadA_f32(A, rowf, ks * 32, half);
    BU Bb[2];
    loadB(Bb[0], ldsW, (ks * 8) * 64, lane);
#pragma unroll
    for (int nt = 0; nt < 8; ++nt) {
      if (nt < 7) loadB(Bb[(nt + 1) & 1], ldsW, (ks * 8 + nt + 1) * 64, lane);
      acc[nt] = __builtin_amdgcn_wmma_f32_16x16x32_bf16(
          false, A.v, false, Bb[nt & 1].v, (short)0, acc[nt], false, false);
    }
  }

  // ---- bias + SiLU, restage as bf16 for layer-2 A fragments ----
  uint16_t* hrow = &h1s[wave * 16 * EMB];
#pragma unroll
  for (int nt = 0; nt < 8; ++nt) {
    float b = b1[nt * 16 + c];
#pragma unroll
    for (int vv = 0; vv < 8; ++vv) {
      float hv = silu(acc[nt][vv] + b);
      hrow[(vv + half * 8) * EMB + nt * 16 + c] = f2bf_fast(hv);
    }
  }
  lds_fence();

  // ---- layer 2: (16x128) @ (128x64) ----
  v8f acc2[4] = {};
  const uint4* hq = (const uint4*)(hrow + c * EMB);  // 256B-aligned row
#pragma unroll
  for (int ks = 0; ks < 4; ++ks) {
    BU A;
    A.q[0] = hq[ks * 4 + half];      // words 0..3
    A.q[1] = hq[ks * 4 + 2 + half];  // words 4..7
    BU Bb[2];
    loadB(Bb[0], ldsW, 2048 + (ks * 4) * 64, lane);
#pragma unroll
    for (int nt = 0; nt < 4; ++nt) {
      if (nt < 3)
        loadB(Bb[(nt + 1) & 1], ldsW, 2048 + (ks * 4 + nt + 1) * 64, lane);
      acc2[nt] = __builtin_amdgcn_wmma_f32_16x16x32_bf16(
          false, A.v, false, Bb[nt & 1].v, (short)0, acc2[nt], false, false);
    }
  }

  // ---- layer 3 dot with W3 + full reduction -> atomic add to prop ----
  float p = 0.0f;
#pragma unroll
  for (int nt = 0; nt < 4; ++nt) {
    float b  = b2[nt * 16 + c];
    float w3 = W3[nt * 16 + c];
#pragma unroll
    for (int vv = 0; vv < 8; ++vv)
      p += silu(acc2[nt][vv] + b) * w3;
  }
  p += __shfl_xor(p, 1, 32);
  p += __shfl_xor(p, 2, 32);
  p += __shfl_xor(p, 4, 32);
  p += __shfl_xor(p, 8, 32);
  p += __shfl_xor(p, 16, 32);
  if (lane == 0) atomicAdd(out, p);
}

// ------------------------------- force head --------------------------------
template <bool XBF>
__global__ __launch_bounds__(256) void force_kernel(
    const float* __restrict__ x, const uint16_t* __restrict__ xbf,
    const int* __restrict__ idx_s, const int* __restrict__ idx_t,
    const float* __restrict__ evst, const float* __restrict__ bf1,
    const float* __restrict__ bf2, const float* __restrict__ Wf3,
    const uint16_t* __restrict__ wblob, float* __restrict__ out) {
  __shared__ uint4 ldsW[5120];                          // 80KB: wf1p + wf2p
  __shared__ __align__(16) uint16_t h1s[8 * 16 * EMB];  // 32KB
  __shared__ float fst[8][16];

  stage_weights((const uint4*)wblob, ldsW, 5120);

  int lane = threadIdx.x & 31;
  int wave = threadIdx.x >> 5;
  int half = lane >> 4;
  int c    = lane & 15;

  // hoist per-lane loop-invariant bias/W3 values
  float bias1[8], bias2[4], w3v[4];
#pragma unroll
  for (int nt = 0; nt < 8; ++nt) bias1[nt] = bf1[nt * 16 + c];
#pragma unroll
  for (int nt = 0; nt < 4; ++nt) {
    bias2[nt] = bf2[nt * 16 + c];
    w3v[nt]   = Wf3[nt * 16 + c];
  }

  for (int tile = blockIdx.x * 8 + wave; tile < NEDGES / 16;
       tile += gridDim.x * 8) {
    int edge = tile * 16 + c;
    int ns  = idx_s[edge];
    int ntg = idx_t[edge];

    // ---- layer 1: (16x256) @ (256x128); K 0..127 from x[s], 128..255 x[t] ----
    v8f acc[8] = {};
#pragma unroll
    for (int ks = 0; ks < 8; ++ks) {
      size_t node = (size_t)((ks < 4) ? ns : ntg);
      int kb = (ks & 3) * 32;
      BU A;
      if (XBF) loadA_bf(A, xbf + node * EMB, kb, half);
      else     loadA_f32(A, x + node * EMB, kb, half);
      BU Bb[2];
      loadB(Bb[0], ldsW, (ks * 8) * 64, lane);
#pragma unroll
      for (int nt = 0; nt < 8; ++nt) {
        if (nt < 7) loadB(Bb[(nt + 1) & 1], ldsW, (ks * 8 + nt + 1) * 64, lane);
        acc[nt] = __builtin_amdgcn_wmma_f32_16x16x32_bf16(
            false, A.v, false, Bb[nt & 1].v, (short)0, acc[nt], false, false);
      }
    }

    // ---- bias + SiLU, restage bf16 ----
    uint16_t* hrow = &h1s[wave * 16 * EMB];
#pragma unroll
    for (int nt = 0; nt < 8; ++nt) {
#pragma unroll
      for (int vv = 0; vv < 8; ++vv) {
        float hv = silu(acc[nt][vv] + bias1[nt]);
        hrow[(vv + half * 8) * EMB + nt * 16 + c] = f2bf_fast(hv);
      }
    }
    lds_fence();

    // ---- layer 2: (16x128) @ (128x64) ----
    v8f acc2[4] = {};
    const uint4* hq = (const uint4*)(hrow + c * EMB);
#pragma unroll
    for (int ks = 0; ks < 4; ++ks) {
      BU A;
      A.q[0] = hq[ks * 4 + half];
      A.q[1] = hq[ks * 4 + 2 + half];
      BU Bb[2];
      loadB(Bb[0], ldsW, 4096 + (ks * 4) * 64, lane);
#pragma unroll
      for (int nt = 0; nt < 4; ++nt) {
        if (nt < 3)
          loadB(Bb[(nt + 1) & 1], ldsW, 4096 + (ks * 4 + nt + 1) * 64, lane);
        acc2[nt] = __builtin_amdgcn_wmma_f32_16x16x32_bf16(
            false, A.v, false, Bb[nt & 1].v, (short)0, acc2[nt], false, false);
      }
    }

    // ---- layer 3 (64->1) + cross-lane reduce: per-edge scalar f ----
    float s[8] = {};
#pragma unroll
    for (int nt = 0; nt < 4; ++nt) {
#pragma unroll
      for (int vv = 0; vv < 8; ++vv)
        s[vv] += silu(acc2[nt][vv] + bias2[nt]) * w3v[nt];
    }
#pragma unroll
    for (int vv = 0; vv < 8; ++vv) {
      float t = s[vv];
      t += __shfl_xor(t, 1, 32);
      t += __shfl_xor(t, 2, 32);
      t += __shfl_xor(t, 4, 32);
      t += __shfl_xor(t, 8, 32);
      s[vv] = t;   // valid where c==0: row m = vv + 8*half
    }
    if (c == 0) {
#pragma unroll
      for (int vv = 0; vv < 8; ++vv) fst[wave][half * 8 + vv] = s[vv];
    }
    lds_fence();

    // ---- scale edge vector, scatter-add into force_s ----
    if (lane < 16) {
      float fv = fst[wave][lane];
      const float* ep = evst + (size_t)edge * 3;   // edge == tile*16+lane here
      float* fo = out + 1 + (size_t)ns * 3;
      atomicAdd(fo + 0, fv * ep[0]);
      atomicAdd(fo + 1, fv * ep[1]);
      atomicAdd(fo + 2, fv * ep[2]);
    }
  }
}

// ------------------------------- launcher ----------------------------------
extern "C" void kernel_launch(void* const* d_in, const int* in_sizes, int n_in,
                              void* d_out, int out_size, void* d_ws, size_t ws_size,
                              hipStream_t stream) {
  const float* x     = (const float*)d_in[0];
  const int*   idx_s = (const int*)d_in[1];
  const int*   idx_t = (const int*)d_in[2];
  const float* evst  = (const float*)d_in[3];
  // d_in[4] = pos (unused by reference outputs)
  const float* W1  = (const float*)d_in[5];
  const float* b1  = (const float*)d_in[6];
  const float* W2  = (const float*)d_in[7];
  const float* b2  = (const float*)d_in[8];
  const float* W3  = (const float*)d_in[9];
  const float* Wf1 = (const float*)d_in[10];
  const float* bf1 = (const float*)d_in[11];
  const float* Wf2 = (const float*)d_in[12];
  const float* bf2 = (const float*)d_in[13];
  const float* Wf3 = (const float*)d_in[14];
  uint16_t* ws  = (uint16_t*)d_ws;
  float*    out = (float*)d_out;

  const size_t kWeightElems = 65536;                       // 128KB of bf16
  const size_t kXElems      = (size_t)NNODES * EMB;        // 6.4M bf16
  bool use_xbf = ws_size >= (kWeightElems + kXElems) * sizeof(uint16_t);
  uint16_t* xbf = ws + kWeightElems;

  zero_out_kernel<<<(out_size + 255) / 256, 256, 0, stream>>>(out, out_size);
  prep_weights_kernel<<<65536 / 256, 256, 0, stream>>>(W1, W2, Wf1, Wf2, ws);

  if (use_xbf) {
    int n2 = (int)(kXElems / 2);
    xcvt_kernel<<<(n2 + 255) / 256, 256, 0, stream>>>(x, (uint32_t*)xbf, n2);
    prop_kernel<true><<<(NNODES / 16 + 7) / 8, 256, 0, stream>>>(
        x, xbf, b1, b2, W3, ws, out);
    force_kernel<true><<<1280, 256, 0, stream>>>(
        x, xbf, idx_s, idx_t, evst, bf1, bf2, Wf3, ws + 24576, out);
  } else {
    prop_kernel<false><<<(NNODES / 16 + 7) / 8, 256, 0, stream>>>(
        x, ws, b1, b2, W3, ws, out);
    force_kernel<false><<<1280, 256, 0, stream>>>(
        x, ws, idx_s, idx_t, evst, bf1, bf2, Wf3, ws + 24576, out);
  }
}
